// YOLOvx_16544214024486
// MI455X (gfx1250) — matrix operand order repaired and checked by reading, not compiled, not run on hardware
//
#include <hip/hip_runtime.h>
#include <math.h>

#define NC 80
#define TOPK 1000
#define NLEV 3
#define SLOT_STRIDE 1024
#define NCAND (NLEV * SLOT_STRIDE)   /* 3072, 1024-padded per level */
#define MAXDET 300
#define NBINS 2048
#define OVF_CAP 4096
#define CONF_THRESH 0.05f
#define NMS_THRESH 0.6f
#define CLS_OFFSET_F 4096.0f

typedef __attribute__((ext_vector_type(16))) _Float16 v16h;
typedef __attribute__((ext_vector_type(8)))  float    v8f;

// order-preserving float->uint (monotone increasing)
__device__ __forceinline__ unsigned f2ord(float f) {
    unsigned b = __float_as_uint(f);
    return (b & 0x80000000u) ? ~b : (b | 0x80000000u);
}
__device__ __forceinline__ float ord2f(unsigned u) {
    unsigned b = (u & 0x80000000u) ? (u & 0x7FFFFFFFu) : ~u;
    return __uint_as_float(b);
}
__device__ __forceinline__ float sigmoidf(float x) { return 1.0f / (1.0f + __expf(-x)); }

// ---------------- init: zero histograms + counters ----------------
__global__ void k_init(unsigned* hist, unsigned* cnt) {
    int t = threadIdx.x;
    for (int i = t; i < NLEV * NBINS; i += 1024) hist[i] = 0u;
    if (t < 6) cnt[t] = 0u;
}

// chunking: 4096 cls elements per block. level0: 2,048,000 (500 chunks),
// level1: 512,000 (125 chunks), level2: 128,000 (32 chunks, last partial)
__device__ __forceinline__ void chunk_map(int chunk, int& lev, long& ebase, long& nel) {
    lev = (chunk < 500) ? 0 : ((chunk < 625) ? 1 : 2);
    int lbase = (lev == 0) ? 0 : ((lev == 1) ? 500 : 625);
    nel = (lev == 0) ? 2048000L : ((lev == 1) ? 512000L : 128000L);
    ebase = (long)(chunk - lbase) * 4096L;
}

// ---------------- pass 1: 2048-bin histogram of logits per level ----------------
__global__ void k_hist(const float* __restrict__ c0, const float* __restrict__ c1,
                       const float* __restrict__ c2, unsigned* __restrict__ ghist) {
    __shared__ unsigned lh[NBINS];
    int t = threadIdx.x;
    for (int i = t; i < NBINS; i += 256) lh[i] = 0u;
    __syncthreads();
    int lev; long ebase, nel;
    chunk_map(blockIdx.x, lev, ebase, nel);
    const float* p = (lev == 0) ? c0 : ((lev == 1) ? c1 : c2);
    for (int k = 0; k < 16; ++k) {
        long e = ebase + t + k * 256;
        if (e < nel) {
            unsigned u = f2ord(p[e]);
            atomicAdd(&lh[u >> 21], 1u);
        }
    }
    __syncthreads();
    unsigned* gh = ghist + lev * NBINS;
    for (int i = t; i < NBINS; i += 256)
        if (lh[i]) atomicAdd(&gh[i], lh[i]);
}

// ---------------- find threshold bin b* per level ----------------
__global__ void k_thresh(const unsigned* __restrict__ ghist, unsigned* __restrict__ thr) {
    int lev = threadIdx.x;
    if (lev < NLEV) {
        const unsigned* h = ghist + lev * NBINS;
        unsigned cum = 0; int b = NBINS - 1;
        for (; b >= 0; --b) { cum += h[b]; if (cum >= TOPK) break; }
        if (b < 0) b = 0;
        thr[lev * 2 + 0] = (unsigned)b;
        thr[lev * 2 + 1] = cum - h[b];   // countAbove (strictly above b*) < TOPK
    }
}

// ---------------- pass 2: compact above-threshold + threshold-bin overflow ----------------
__global__ void k_compact(const float* __restrict__ c0, const float* __restrict__ c1,
                          const float* __restrict__ c2, const unsigned* __restrict__ thr,
                          unsigned* __restrict__ cnt, float* __restrict__ candS,
                          unsigned* __restrict__ candI,
                          unsigned long long* __restrict__ ovf) {
    int t = threadIdx.x;
    int lev; long ebase, nel;
    chunk_map(blockIdx.x, lev, ebase, nel);
    const float* p = (lev == 0) ? c0 : ((lev == 1) ? c1 : c2);
    unsigned bstar = thr[lev * 2];
    for (int k = 0; k < 16; ++k) {
        long e = ebase + t + k * 256;
        if (e < nel) {
            float x = p[e];
            unsigned u = f2ord(x);
            unsigned bin = u >> 21;
            if (bin > bstar) {
                unsigned pos = atomicAdd(&cnt[lev], 1u);
                if (pos < (unsigned)SLOT_STRIDE) {
                    candS[lev * SLOT_STRIDE + pos] = sigmoidf(x);
                    candI[lev * SLOT_STRIDE + pos] = (unsigned)e;
                }
            } else if (bin == bstar) {
                unsigned pos = atomicAdd(&cnt[NLEV + lev], 1u);
                if (pos < (unsigned)OVF_CAP)
                    ovf[lev * OVF_CAP + pos] =
                        (((unsigned long long)u) << 32) | (unsigned)(~(unsigned)e);
            }
        }
    }
}

// ---------------- resolve threshold-bin ties deterministically (bitonic sort) ----------------
__global__ void k_tie(const unsigned* __restrict__ thr, const unsigned* __restrict__ cnt,
                      const unsigned long long* __restrict__ ovf,
                      float* __restrict__ candS, unsigned* __restrict__ candI) {
    __shared__ unsigned long long key[OVF_CAP];
    int lev = blockIdx.x, t = threadIdx.x, nt = blockDim.x;
    unsigned ca = thr[lev * 2 + 1];
    unsigned c = cnt[NLEV + lev];
    int n = (c < (unsigned)OVF_CAP) ? (int)c : OVF_CAP;
    for (int i = t; i < OVF_CAP; i += nt) key[i] = (i < n) ? ovf[lev * OVF_CAP + i] : 0ULL;
    __syncthreads();
    for (int k = 2; k <= OVF_CAP; k <<= 1) {
        for (int j = k >> 1; j > 0; j >>= 1) {
            for (int i = t; i < OVF_CAP; i += nt) {
                int ix = i ^ j;
                if (ix > i) {
                    bool up = ((i & k) == 0);              // descending segments
                    unsigned long long a = key[i], b = key[ix];
                    bool sw = up ? (a < b) : (a > b);
                    if (sw) { key[i] = b; key[ix] = a; }
                }
            }
            __syncthreads();
        }
    }
    int rem = TOPK - (int)ca;                               // >= 1 by construction
    for (int i = t; i < rem; i += nt) {
        float s; unsigned idx;
        if (i < n) {
            unsigned long long kk = key[i];
            unsigned u = (unsigned)(kk >> 32);
            idx = ~(unsigned)(kk & 0xFFFFFFFFu);
            s = sigmoidf(ord2f(u));
        } else { s = -__builtin_inff(); idx = 0u; }
        candS[lev * SLOT_STRIDE + ca + i] = s;
        candI[lev * SLOT_STRIDE + ca + i] = idx;
    }
}

// ---------------- DFL decode via WMMA: 1 wave = 16 candidates, 2 wmma ops ----------------
__global__ void __launch_bounds__(32)
k_decode(const float* __restrict__ r0, const float* __restrict__ r1,
         const float* __restrict__ r2, const float* __restrict__ candS,
         const unsigned* __restrict__ candI, float* __restrict__ boxes,
         float* __restrict__ nb, float* __restrict__ areas, float* __restrict__ live,
         float* __restrict__ gscore, int* __restrict__ glabel) {
    int lane = threadIdx.x;          // 0..31, single wave
    int c0 = blockIdx.x * 16;        // global slot base (16 | 1024 -> no level straddle)
    int lev = c0 / SLOT_STRIDE;
    const float* regp = (lev == 0) ? r0 : ((lev == 1) ? r1 : r2);
    int w = (lev == 0) ? 160 : ((lev == 1) ? 80 : 40);
    float stride = (lev == 0) ? 8.f : ((lev == 1) ? 16.f : 32.f);

    int m = lane & 15;               // candidate (A-matrix row M)
    int s = c0 + m;
    int ls = s - lev * SLOT_STRIDE;  // local slot, < 1024
    bool slotOK = (ls < TOPK);
    unsigned idx = candI[lev * SLOT_STRIDE + ls];
    if (!slotOK) idx = 0u;
    float score = slotOK ? candS[lev * SLOT_STRIDE + ls] : -__builtin_inff();
    unsigned anchor = idx / NC;
    const float* base = regp + (size_t)anchor * 64;   // [4 sides][16 bins]
    int kof = (lane < 16) ? 0 : 8;   // A layout: low half-wave owns bins 0-7, high owns 8-15

    float v[4][8];
#pragma unroll
    for (int sd = 0; sd < 4; ++sd)
#pragma unroll
        for (int j = 0; j < 8; ++j) v[sd][j] = base[sd * 16 + kof + j];

    float mx[4];
#pragma unroll
    for (int sd = 0; sd < 4; ++sd) {
        float ml = v[sd][0];
#pragma unroll
        for (int j = 1; j < 8; ++j) ml = fmaxf(ml, v[sd][j]);
        mx[sd] = fmaxf(ml, __shfl_xor(ml, 16, 32));  // combine across half-waves
    }

    // A per 16-bit 16x32 layout: lane<16 halves0-7 -> K=0-7, halves8-15 -> K=16-23;
    // lane>=16 halves0-7 -> K=8-15, halves8-15 -> K=24-31.  K = side*16 + bin.
    v16h A1, A2, B;
#pragma unroll
    for (int h = 0; h < 8; ++h) {
        A1[h]     = (_Float16)__expf(v[0][h] - mx[0]);
        A1[h + 8] = (_Float16)__expf(v[1][h] - mx[1]);
        A2[h]     = (_Float16)__expf(v[2][h] - mx[2]);
        A2[h + 8] = (_Float16)__expf(v[3][h] - mx[3]);
    }
    // B 32x16: lane = column N (mod 16), same K order as A: side = (h >= 8), bin = (h & 7) + (lane<16 ? 0 : 8).
    // col0=[proj;0] col1=[1;0] col2=[0;proj] col3=[0;1] -> D[m][0..3] = dot0,sum0,dot1,sum1
    int N = lane & 15; bool lo = (lane < 16);
#pragma unroll
    for (int h = 0; h < 16; ++h) {
        float pv = (16.0f / 15.0f) * (float)((h & 7) + (lo ? 0 : 8));   // jnp.linspace(0,16,16)
        float bv = 0.f;
        if (N == 0 && h < 8)  bv = pv;
        else if (N == 1 && h < 8)  bv = 1.f;
        else if (N == 2 && h >= 8) bv = pv;
        else if (N == 3 && h >= 8) bv = 1.f;
        B[h] = (_Float16)bv;
    }

    v8f Cz = {};
    v8f D1 = __builtin_amdgcn_wmma_f32_16x16x32_f16(false, A1, false, B, (short)0, Cz, false, false);
    v8f D2 = __builtin_amdgcn_wmma_f32_16x16x32_f16(false, A2, false, B, (short)0, Cz, false, false);

    // D layout: VGPR r -> row M=r (lanes 0-15) / M=r+8 (lanes 16-31), col N=lane%16
    __shared__ float L1[16][16], L2[16][16];
    int row0 = (lane < 16) ? 0 : 8, col = lane & 15;
#pragma unroll
    for (int r = 0; r < 8; ++r) { L1[row0 + r][col] = D1[r]; L2[row0 + r][col] = D2[r]; }
    __syncthreads();

    if (lane < 16) {
        float d0 = L1[m][0] / L1[m][1];
        float d1 = L1[m][2] / L1[m][3];
        float d2 = L2[m][0] / L2[m][1];
        float d3 = L2[m][2] / L2[m][3];
        float ax = ((float)(anchor % (unsigned)w) + 0.5f) * stride;
        float ay = ((float)(anchor / (unsigned)w) + 0.5f) * stride;
        float x1 = ax - d0 * stride, y1 = ay - d1 * stride;
        float x2 = ax + d2 * stride, y2 = ay + d3 * stride;
        int label = (int)(idx % NC);
        if (!slotOK) { x1 = y1 = x2 = y2 = 0.f; label = 0; }
        float off = (float)label * CLS_OFFSET_F;
        boxes[s * 4 + 0] = x1; boxes[s * 4 + 1] = y1;
        boxes[s * 4 + 2] = x2; boxes[s * 4 + 3] = y2;
        nb[s * 4 + 0] = x1 + off; nb[s * 4 + 1] = y1 + off;
        nb[s * 4 + 2] = x2 + off; nb[s * 4 + 3] = y2 + off;
        areas[s] = (x2 - x1) * (y2 - y1);
        gscore[s] = slotOK ? score : 0.f;
        glabel[s] = label;
        live[s] = (score > CONF_THRESH) ? score : -__builtin_inff();
    }
}

// ---------------- greedy NMS (single workgroup, LDS-resident live scores) ----------------
__global__ void __launch_bounds__(1024)
k_nms(const float* __restrict__ boxes, const float* __restrict__ nb,
      const float* __restrict__ areas, const float* __restrict__ live_in,
      const float* __restrict__ gscore, const int* __restrict__ glabel,
      float* __restrict__ out) {
    __shared__ float live[NCAND];
    __shared__ float rv[1024]; __shared__ int ri[1024];
    __shared__ int selI[MAXDET]; __shared__ float selS[MAXDET];
    __shared__ int bI;
    int t = threadIdx.x;
    for (int i = t; i < NCAND; i += 1024) live[i] = live_in[i];
    __syncthreads();
    for (int it = 0; it < MAXDET; ++it) {
        // argmax with lowest-index tie-break (matches jnp.argmax, incl. all -inf -> 0)
        float bv = -__builtin_inff(); int bi = 0x7FFFFFFF;
        for (int i = t; i < NCAND; i += 1024) {
            float v = live[i];
            if (v > bv || (v == bv && i < bi)) { bv = v; bi = i; }
        }
        rv[t] = bv; ri[t] = bi;
        __syncthreads();
        for (int sft = 512; sft > 0; sft >>= 1) {
            if (t < sft) {
                float ov = rv[t + sft]; int oi = ri[t + sft];
                if (ov > rv[t] || (ov == rv[t] && oi < ri[t])) { rv[t] = ov; ri[t] = oi; }
            }
            __syncthreads();
        }
        if (t == 0) { bI = ri[0]; selI[it] = ri[0]; selS[it] = rv[0]; }
        __syncthreads();
        int i = bI;
        float x1i = nb[i * 4 + 0], y1i = nb[i * 4 + 1];
        float x2i = nb[i * 4 + 2], y2i = nb[i * 4 + 3], ai = areas[i];
        for (int j = t; j < NCAND; j += 1024) {
            float iw = fminf(nb[j * 4 + 2], x2i) - fmaxf(nb[j * 4 + 0], x1i);
            float ih = fminf(nb[j * 4 + 3], y2i) - fmaxf(nb[j * 4 + 1], y1i);
            iw = fmaxf(iw, 0.f); ih = fmaxf(ih, 0.f);
            float inter = iw * ih;
            float iou = inter / (areas[j] + ai - inter + 1e-10f);
            if (iou > NMS_THRESH) live[j] = -__builtin_inff();
        }
        __syncthreads();
    }
    // outputs: boxes[300*4] | scores[300] | labels[300] (labels as float values)
    for (int tt = t; tt < MAXDET; tt += 1024) {
        int i = selI[tt]; float si = selS[tt];
        bool valid = (si > CONF_THRESH);   // -inf fails
        for (int k = 0; k < 4; ++k) out[tt * 4 + k] = valid ? boxes[i * 4 + k] : 0.f;
        out[1200 + tt] = valid ? gscore[i] : 0.f;
        out[1500 + tt] = valid ? (float)glabel[i] : -1.0f;
    }
}

extern "C" void kernel_launch(void* const* d_in, const int* in_sizes, int n_in,
                              void* d_out, int out_size, void* d_ws, size_t ws_size,
                              hipStream_t stream) {
    (void)in_sizes; (void)n_in; (void)out_size; (void)ws_size;
    // dict order: cls_p3, reg_p3, cls_p4, reg_p4, cls_p5, reg_p5 ; only batch 0 is used.
    const float* cls0 = (const float*)d_in[0];
    const float* reg0 = (const float*)d_in[1];
    const float* cls1 = (const float*)d_in[2];
    const float* reg1 = (const float*)d_in[3];
    const float* cls2 = (const float*)d_in[4];
    const float* reg2 = (const float*)d_in[5];
    float* out = (float*)d_out;

    char* ws = (char*)d_ws;
    size_t off = 0;
    auto alloc = [&](size_t b) { size_t o = off; off = (off + b + 255) & ~(size_t)255; return o; };
    unsigned* hist            = (unsigned*)(ws + alloc((size_t)NLEV * NBINS * 4));
    unsigned* cnt             = (unsigned*)(ws + alloc(6 * 4));
    unsigned* thr             = (unsigned*)(ws + alloc(6 * 4));
    float*    candS           = (float*)(ws + alloc((size_t)NCAND * 4));
    unsigned* candI           = (unsigned*)(ws + alloc((size_t)NCAND * 4));
    unsigned long long* ovf   = (unsigned long long*)(ws + alloc((size_t)NLEV * OVF_CAP * 8));
    float* boxes              = (float*)(ws + alloc((size_t)NCAND * 16));
    float* nbx                = (float*)(ws + alloc((size_t)NCAND * 16));
    float* areas              = (float*)(ws + alloc((size_t)NCAND * 4));
    float* live               = (float*)(ws + alloc((size_t)NCAND * 4));
    float* gscore             = (float*)(ws + alloc((size_t)NCAND * 4));
    int*   glabel             = (int*)(ws + alloc((size_t)NCAND * 4));

    k_init   <<<1, 1024, 0, stream>>>(hist, cnt);
    k_hist   <<<657, 256, 0, stream>>>(cls0, cls1, cls2, hist);
    k_thresh <<<1, 32, 0, stream>>>(hist, thr);
    k_compact<<<657, 256, 0, stream>>>(cls0, cls1, cls2, thr, cnt, candS, candI, ovf);
    k_tie    <<<NLEV, 1024, 0, stream>>>(thr, cnt, ovf, candS, candI);
    k_decode <<<NCAND / 16, 32, 0, stream>>>(reg0, reg1, reg2, candS, candI,
                                             boxes, nbx, areas, live, gscore, glabel);
    k_nms    <<<1, 1024, 0, stream>>>(boxes, nbx, areas, live, gscore, glabel, out);
}
